// SolarRecommendationGNN_22239340659075
// MI455X (gfx1250) — matrix operand order repaired and checked by reading, not compiled, not run on hardware
//
#include <hip/hip_runtime.h>
#include <hip/hip_bf16.h>

#define H 128
#define TSTEPS 24
#define GROUPS 15
#define KCLS 15
#define NT 4   // 16x16 N-tiles per wave -> 64 cols per wave

typedef __attribute__((ext_vector_type(16))) _Float16 v16h;
typedef __attribute__((ext_vector_type(8)))  _Float16 v8h;
typedef __attribute__((ext_vector_type(8)))  float    v8f;

// ---------------- WMMA GEMM helpers ----------------

// A tile fragment: lane = M row (lane&15), hh = lane>>4 selects K-halves.
// Per ISA 7.12.2: elements 0..7  <-> K = k0 + hh*8 + 0..7   (contiguous)
//                 elements 8..15 <-> K = k0 + 16 + hh*8 + 0..7 (contiguous)
static __device__ __forceinline__ v16h load_a_f32(const float* __restrict__ p, int hh, int prerelu) {
  const float4* q0 = (const float4*)(p + hh * 8);
  const float4* q1 = (const float4*)(p + 16 + hh * 8);
  float4 f0 = q0[0], f1 = q0[1], f2 = q1[0], f3 = q1[1];
  float vals[16] = {f0.x, f0.y, f0.z, f0.w, f1.x, f1.y, f1.z, f1.w,
                    f2.x, f2.y, f2.z, f2.w, f3.x, f3.y, f3.z, f3.w};
  v16h a;
#pragma unroll
  for (int i = 0; i < 16; ++i) {
    float v = vals[i];
    if (prerelu) v = fmaxf(v, 0.0f);
    a[i] = (_Float16)v;
  }
  return a;
}

// B tile fragment: lane = N col (lane&15); elements 0..7 <-> K = k0 + hh*8 + 0..7,
// elements 8..15 <-> K = k0 + 16 + hh*8 + 0..7 of row-major W[K,N] (stride ldn).
static __device__ __forceinline__ v16h load_b_f16(const _Float16* __restrict__ p, int ldn, int hh) {
  v16h b;
#pragma unroll
  for (int i = 0; i < 8; ++i) { b[i] = p[(size_t)(hh * 8 + i) * ldn]; b[8 + i] = p[(size_t)(16 + hh * 8 + i) * ldn]; }
  return b;
}

// C = act( A0[M,K0] @ W0[K0,N] (+ A1[M,K1] @ W1[K1,N]) + bias ), fp32 out.
// One wave (32 lanes) per 16-row strip x 4 n-tiles. M must be a multiple of 16.
__global__ void gemm_wmma_k(const float* __restrict__ A0, int lda0, int K0,
                            const _Float16* __restrict__ W0,
                            const float* __restrict__ A1, int lda1, int K1,
                            const _Float16* __restrict__ W1,
                            const float* __restrict__ bias,
                            float* __restrict__ C, int N, int M,
                            int prerelu, int postrelu) {
  int lane  = threadIdx.x;                       // 0..31
  int mtile = blockIdx.x * blockDim.y + threadIdx.y;
  if (mtile * 16 >= M) return;                   // whole-wave uniform skip (EXEC stays all-1)
  int nbase = blockIdx.y * (16 * NT);
  int hh = lane >> 4;
  int ml = lane & 15;
  int row = mtile * 16 + ml;

  v8f acc[NT];
#pragma unroll
  for (int t = 0; t < NT; ++t)
#pragma unroll
    for (int r = 0; r < 8; ++r) acc[t][r] = 0.0f;

  const float* a0row = A0 + (size_t)row * lda0;
  for (int k0 = 0; k0 < K0; k0 += 32) {
    v16h a = load_a_f32(a0row + k0, hh, prerelu);
    const _Float16* wrow = W0 + (size_t)k0 * N + nbase + ml;
#pragma unroll
    for (int t = 0; t < NT; ++t) {
      v16h b = load_b_f16(wrow + t * 16, N, hh);
      acc[t] = __builtin_amdgcn_wmma_f32_16x16x32_f16(false, a, false, b,
                                                      (short)0, acc[t], false, false);
    }
  }
  if (A1) {
    const float* a1row = A1 + (size_t)row * lda1;
    for (int k0 = 0; k0 < K1; k0 += 32) {
      v16h a = load_a_f32(a1row + k0, hh, prerelu);
      const _Float16* wrow = W1 + (size_t)k0 * N + nbase + ml;
#pragma unroll
      for (int t = 0; t < NT; ++t) {
        v16h b = load_b_f16(wrow + t * 16, N, hh);
        acc[t] = __builtin_amdgcn_wmma_f32_16x16x32_f16(false, a, false, b,
                                                        (short)0, acc[t], false, false);
      }
    }
  }
  // D layout: VGPR r -> row = mtile*16 + r + hh*8, col = nbase + t*16 + (lane&15)
#pragma unroll
  for (int t = 0; t < NT; ++t) {
    int n = nbase + t * 16 + ml;
    float bv = bias ? bias[n] : 0.0f;
#pragma unroll
    for (int r = 0; r < 8; ++r) {
      int m = mtile * 16 + r + hh * 8;
      float v = acc[t][r] + bv;
      if (postrelu) v = fmaxf(v, 0.0f);
      C[(size_t)m * N + n] = v;
    }
  }
}

// ---------------- small utility kernels ----------------

__global__ void cvt_h_k(const float* __restrict__ s, _Float16* __restrict__ d, int cnt) {
  int i = blockIdx.x * blockDim.x + threadIdx.x;
  if (i < cnt) d[i] = (_Float16)s[i];
}

__global__ void fill_k(float* __restrict__ p, float v, int cnt) {
  int i = blockIdx.x * blockDim.x + threadIdx.x;
  if (i < cnt) p[i] = v;
}

__global__ void bias_sum_k(const float* __restrict__ a, const float* __restrict__ b,
                           float* __restrict__ o, int cnt) {
  int i = blockIdx.x * blockDim.x + threadIdx.x;
  if (i < cnt) o[i] = a[i] + b[i];
}

__global__ void deg_edge_k(const int* __restrict__ dst, float* __restrict__ deg, int E) {
  int e = blockIdx.x * blockDim.x + threadIdx.x;
  if (e < E) atomicAdd(&deg[dst[e]], 1.0f);
}

__global__ void dinv_k(const float* __restrict__ deg, float* __restrict__ dinv, int N) {
  int n = blockIdx.x * blockDim.x + threadIdx.x;
  if (n < N) dinv[n] = rsqrtf(fmaxf(deg[n], 1.0f));
}

// agg = gcn_b[f] + dinv[n]^2 * xw[n,f]  (self-loop contribution + bias)
__global__ void agg_init_k(const float* __restrict__ xw, const float* __restrict__ dinv,
                           const float* __restrict__ gb, float* __restrict__ agg, int N) {
  int i = blockIdx.x * blockDim.x + threadIdx.x;
  if (i >= N * H) return;
  int n = i >> 7, f = i & (H - 1);
  float di = dinv[n];
  agg[i] = gb[f] + di * di * xw[i];
}

// edge scatter: agg[dst] += xw[src] * dinv[src]*dinv[dst]; 4 features/thread
__global__ void edge_scatter_k(const int* __restrict__ src, const int* __restrict__ dst,
                               const float* __restrict__ dinv, const float* __restrict__ xw,
                               float* __restrict__ agg, int E) {
  int idx = blockIdx.x * blockDim.x + threadIdx.x;
  if (idx >= E * 32) return;
  int e = idx >> 5;
  int f = (idx & 31) * 4;
  int s = src[e], d = dst[e];
  float w = dinv[s] * dinv[d];
  float4 v = *(const float4*)(xw + (size_t)s * H + f);
  float* ap = agg + (size_t)d * H + f;
  atomicAdd(ap + 0, v.x * w);
  atomicAdd(ap + 1, v.y * w);
  atomicAdd(ap + 2, v.z * w);
  atomicAdd(ap + 3, v.w * w);
}

// LV constraint: G sequential conditional L2-normalizations, then residual.
// One wave32 per node; each lane owns 4 of the 128 features.
__global__ void lv_norm_k(const float* __restrict__ hc, const int* __restrict__ gmask,
                          float* __restrict__ h, int N, int add) {
  int lane = threadIdx.x;
  int n = blockIdx.x * blockDim.y + threadIdx.y;
  if (n >= N) return;
  float4 v = *(const float4*)(hc + (size_t)n * H + lane * 4);
  const int* mrow = gmask + (size_t)n * GROUPS;
  for (int g = 0; g < GROUPS; ++g) {
    float ss = v.x * v.x + v.y * v.y + v.z * v.z + v.w * v.w;
#pragma unroll
    for (int m = 16; m >= 1; m >>= 1) ss += __shfl_xor(ss, m, 32);
    if (mrow[g] > 0) {
      float s = 1.0f / fmaxf(sqrtf(ss), 1e-12f);
      v.x *= s; v.y *= s; v.z *= s; v.w *= s;
    }
  }
  float* hp = h + (size_t)n * H + lane * 4;
  if (add) {
    float4 p = *(const float4*)hp;
    v.x += p.x; v.y += p.y; v.z += p.z; v.w += p.w;
  }
  *(float4*)hp = v;
}

__device__ __forceinline__ float sigm(float x) { return 1.0f / (1.0f + expf(-x)); }

// gates layout per row: [i(0:128) | f(128:256) | g(256:384) | o(384:512)]
__global__ void lstm_cell_k(const float* __restrict__ gates, float* __restrict__ c,
                            float* __restrict__ ht, int N) {
  int i = blockIdx.x * blockDim.x + threadIdx.x;
  if (i >= N * H) return;
  int n = i >> 7, f = i & (H - 1);
  const float* g = gates + (size_t)n * (4 * H);
  float ig = g[f], fg = g[H + f], gg = g[2 * H + f], og = g[3 * H + f];
  float cn = sigm(fg) * c[i] + sigm(ig) * tanhf(gg);
  c[i] = cn;
  ht[i] = sigm(og) * tanhf(cn);
}

__global__ void add_k(float* __restrict__ a, const float* __restrict__ b, int cnt) {
  int i = blockIdx.x * blockDim.x + threadIdx.x;
  if (i < cnt) a[i] += b[i];
}

__global__ void meanh_k(const float* __restrict__ h, float* __restrict__ meanh, int N) {
  int f = blockIdx.x, tid = threadIdx.x;
  float s = 0.0f;
  for (int n = tid; n < N; n += blockDim.x) s += h[(size_t)n * H + f];
  __shared__ float sm[256];
  sm[tid] = s;
  __syncthreads();
  for (int o = 128; o > 0; o >>= 1) {
    if (tid < o) sm[tid] += sm[tid + o];
    __syncthreads();
  }
  if (tid == 0) meanh[f] = sm[0] / (float)N;
}

// mt[j] = sh_b1[j] + sum_k meanh[k] * sh_w1[(128+k)*128 + j]
__global__ void meanterm_k(const float* __restrict__ meanh, const float* __restrict__ sh_w1,
                           const float* __restrict__ sh_b1, float* __restrict__ mt) {
  int j = threadIdx.x;
  float s = sh_b1[j];
  for (int k = 0; k < H; ++k) s += meanh[k] * sh_w1[(size_t)(H + k) * H + j];
  mt[j] = s;
}

__global__ void cl_head_k(const float* __restrict__ chh, const float* __restrict__ w2,
                          const float* __restrict__ b2, float* __restrict__ cl, int N) {
  int n = blockIdx.x * blockDim.y + threadIdx.y;
  int k = threadIdx.x;
  if (n >= N || k >= KCLS) return;
  const float* r = chh + (size_t)n * 64;
  float s = b2[k];
  for (int j = 0; j < 64; ++j) s += r[j] * w2[j * KCLS + k];
  cl[(size_t)n * KCLS + k] = s;
}

__global__ void softmax15_k(const float* __restrict__ cl, float* __restrict__ cp, int N) {
  int n = blockIdx.x * blockDim.x + threadIdx.x;
  if (n >= N) return;
  const float* r = cl + (size_t)n * KCLS;
  float mx = r[0];
  for (int k = 1; k < KCLS; ++k) mx = fmaxf(mx, r[k]);
  float e[KCLS], s = 0.0f;
  for (int k = 0; k < KCLS; ++k) { e[k] = expf(r[k] - mx); s += e[k]; }
  float inv = 1.0f / s;
  float* o = cp + (size_t)n * KCLS;
  for (int k = 0; k < KCLS; ++k) o[k] = e[k] * inv;
}

__global__ void ss_head_k(const float* __restrict__ shh, const float* __restrict__ w2,
                          const float* __restrict__ b2, float* __restrict__ ss, int N) {
  int n = blockIdx.x * blockDim.x + threadIdx.x;
  if (n >= N) return;
  const float* r = shh + (size_t)n * H;
  float s = b2[0];
  for (int j = 0; j < H; ++j) s += r[j] * w2[j];
  ss[n] = sigm(s);
}

// ---------------- launcher ----------------

extern "C" void kernel_launch(void* const* d_in, const int* in_sizes, int n_in,
                              void* d_out, int out_size, void* d_ws, size_t ws_size,
                              hipStream_t stream) {
  const float* x        = (const float*)d_in[0];
  const int*   ei       = (const int*)d_in[1];
  const float* temporal = (const float*)d_in[2];
  const int*   gmask    = (const int*)d_in[3];
  const float* enc_w1   = (const float*)d_in[4];
  const float* enc_b1   = (const float*)d_in[5];
  const float* enc_w2   = (const float*)d_in[6];
  const float* enc_b2   = (const float*)d_in[7];
  const float* gcn_w    = (const float*)d_in[8];
  const float* gcn_b    = (const float*)d_in[9];
  const float* lv_w     = (const float*)d_in[10];
  const float* lv_b     = (const float*)d_in[11];
  const float* wih      = (const float*)d_in[12];
  const float* whh      = (const float*)d_in[13];
  const float* bih      = (const float*)d_in[14];
  const float* bhh      = (const float*)d_in[15];
  const float* ch_w1    = (const float*)d_in[16];
  const float* ch_b1    = (const float*)d_in[17];
  const float* ch_w2    = (const float*)d_in[18];
  const float* ch_b2    = (const float*)d_in[19];
  const float* sh_w1    = (const float*)d_in[20];
  const float* sh_b1    = (const float*)d_in[21];
  const float* sh_w2    = (const float*)d_in[22];
  const float* sh_b2    = (const float*)d_in[23];

  const int N = in_sizes[0] / H;   // 30000 (multiple of 16)
  const int E = in_sizes[1] / 2;   // 960000
  const int* esrc = ei;
  const int* edst = ei + E;

  float* out = (float*)d_out;
  float* cl  = out;
  float* cp  = out + (size_t)N * KCLS;
  float* ssv = out + (size_t)N * KCLS * 2;
  float* h   = out + (size_t)N * KCLS * 2 + N;   // h lives directly in its d_out slice

  // workspace carve
  char* wsb = (char*)d_ws;
  size_t off = 0;
  auto allocf = [&](size_t n) { float* p = (float*)(wsb + off); off += n * sizeof(float); return p; };
  float* deg   = allocf(N);
  float* dinv  = allocf(N);
  float* bufA  = allocf((size_t)N * H);        // xw / enc hidden / chh / shh
  float* ht    = allocf((size_t)N * H);
  float* cst   = allocf((size_t)N * H);
  float* gates = allocf((size_t)N * 4 * H);    // reused as agg/hc during GCN
  float* meanh = allocf(H);
  float* mt    = allocf(H);
  float* bsum  = allocf(4 * H);
  off = (off + 255) & ~(size_t)255;
  auto alloch = [&](size_t n) { _Float16* p = (_Float16*)(wsb + off); off += n * sizeof(_Float16); return p; };
  _Float16* enc_w1h = alloch(H * H);
  _Float16* enc_w2h = alloch(H * H);
  _Float16* gcn_wh  = alloch(3 * H * H);
  _Float16* lv_wh   = alloch(H * H);
  _Float16* wih_h   = alloch(H * 4 * H);
  _Float16* whh_h   = alloch(H * 4 * H);
  _Float16* ch_w1h  = alloch(H * 64);
  _Float16* sh_w1ah = alloch(H * H);           // first 128 rows of sh_w1

  float* agg = gates;                 // [N,H]
  float* hc  = gates + (size_t)N * H; // [N,H]

  auto cvt = [&](const float* s, _Float16* d, int cnt) {
    cvt_h_k<<<(cnt + 255) / 256, 256, 0, stream>>>(s, d, cnt);
  };
  dim3 gblk(32, 4);
  auto gemm = [&](const float* A0, int lda0, int K0, const _Float16* W0,
                  const float* A1, int lda1, int K1, const _Float16* W1,
                  const float* bias, float* C, int Ncols, int prer, int postr) {
    dim3 grid((N / 16 + 3) / 4, Ncols / (16 * NT));
    gemm_wmma_k<<<grid, gblk, 0, stream>>>(A0, lda0, K0, W0, A1, lda1, K1, W1,
                                           bias, C, Ncols, N, prer, postr);
  };

  // ---- weight prep (f16) ----
  cvt(enc_w1, enc_w1h, H * H);
  cvt(enc_w2, enc_w2h, H * H);
  cvt(gcn_w, gcn_wh, 3 * H * H);
  cvt(lv_w, lv_wh, H * H);
  cvt(wih, wih_h, H * 4 * H);
  cvt(whh, whh_h, H * 4 * H);
  cvt(ch_w1, ch_w1h, H * 64);
  cvt(sh_w1, sh_w1ah, H * H);
  bias_sum_k<<<(4 * H + 255) / 256, 256, 0, stream>>>(bih, bhh, bsum, 4 * H);

  // ---- GCN normalization ----
  fill_k<<<(N + 255) / 256, 256, 0, stream>>>(deg, 1.0f, N);         // self-loops
  deg_edge_k<<<(E + 255) / 256, 256, 0, stream>>>(edst, deg, E);
  dinv_k<<<(N + 255) / 256, 256, 0, stream>>>(deg, dinv, N);

  // ---- encoder ----
  gemm(x, H, H, enc_w1h, nullptr, 0, 0, nullptr, enc_b1, bufA, H, 0, 1);
  gemm(bufA, H, H, enc_w2h, nullptr, 0, 0, nullptr, enc_b2, h, H, 0, 0);

  // ---- 3 GCN layers ----
  for (int i = 0; i < 3; ++i) {
    gemm(h, H, H, gcn_wh + (size_t)i * H * H, nullptr, 0, 0, nullptr, nullptr, bufA, H, 0, 0);
    agg_init_k<<<(N * H + 255) / 256, 256, 0, stream>>>(bufA, dinv, gcn_b + i * H, agg, N);
    edge_scatter_k<<<(E * 32 + 255) / 256, 256, 0, stream>>>(esrc, edst, dinv, bufA, agg, E);
    // hc = relu(agg) @ lv_w + lv_b  (prerelu fuses h_new = relu(agg))
    gemm(agg, H, H, lv_wh, nullptr, 0, 0, nullptr, lv_b, hc, H, 1, 0);
    lv_norm_k<<<(N + 7) / 8, dim3(32, 8), 0, stream>>>(hc, gmask, h, N, i > 0 ? 1 : 0);
  }

  // ---- LSTM (24 steps): gates = x_t@Wih + h_t@Whh + (bih+bhh) ----
  fill_k<<<(N * H + 255) / 256, 256, 0, stream>>>(ht, 0.0f, N * H);
  fill_k<<<(N * H + 255) / 256, 256, 0, stream>>>(cst, 0.0f, N * H);
  for (int t = 0; t < TSTEPS; ++t) {
    gemm(temporal + (size_t)t * H, TSTEPS * H, H, wih_h,
         ht, H, H, whh_h, bsum, gates, 4 * H, 0, 0);
    lstm_cell_k<<<(N * H + 255) / 256, 256, 0, stream>>>(gates, cst, ht, N);
  }
  add_k<<<(N * H + 255) / 256, 256, 0, stream>>>(h, ht, N * H);

  // ---- cluster head ----
  gemm(h, H, H, ch_w1h, nullptr, 0, 0, nullptr, ch_b1, bufA, 64, 0, 1);
  cl_head_k<<<(N + 15) / 16, dim3(16, 16), 0, stream>>>(bufA, ch_w2, ch_b2, cl, N);
  softmax15_k<<<(N + 255) / 256, 256, 0, stream>>>(cl, cp, N);

  // ---- share head (mean term folded into GEMM bias) ----
  meanh_k<<<H, 256, 0, stream>>>(h, meanh, N);
  meanterm_k<<<1, H, 0, stream>>>(meanh, sh_w1, sh_b1, mt);
  gemm(h, H, H, sh_w1ah, nullptr, 0, 0, nullptr, mt, bufA, H, 0, 1);
  ss_head_k<<<(N + 255) / 256, 256, 0, stream>>>(bufA, sh_w2, sh_b2, ssv, N);
}
